// GATv2Block_67173288509941
// MI455X (gfx1250) — matrix operand order repaired and checked by reading, not compiled, not run on hardware
//
#include <hip/hip_runtime.h>

#define NN 50000
#define NE 500000
#define EE 550000            // NE + NN self-loops
#define F  128
#define EA 16
#define OUT_H (NN * F)       // 6,400,000 floats, then 1,000,000 int32 edge_index
#define MTILES (NN / 16)     // 3125, exact

typedef __attribute__((ext_vector_type(16))) _Float16 v16h;
typedef __attribute__((ext_vector_type(8)))  float    v8f;

__device__ __forceinline__ float lrelu(float x, float s) { return x > 0.f ? x : s * x; }

__device__ __forceinline__ void atomicMaxF(float* addr, float val) {
  unsigned* ua = (unsigned*)addr;
  unsigned old = __float_as_uint(*addr);
  while (__uint_as_float(old) < val) {
    unsigned assumed = old;
    old = atomicCAS(ua, assumed, __float_as_uint(val));
    if (old == assumed) break;
  }
}

// ---- self-loop attr: deg + attr segment sums ------------------------------
__global__ void k_prep_zero(float* deg, float* attrsum) {
  int i = blockIdx.x * 256 + threadIdx.x;
  if (i < NN * EA) attrsum[i] = 0.f;
  if (i < NN) deg[i] = 0.f;
}

__global__ void k_deg_attr(const int* __restrict__ ei, const float* __restrict__ eattr,
                           float* deg, float* attrsum) {
  int i = blockIdx.x * 256 + threadIdx.x;
  if (i >= NE * EA) return;
  int e = i >> 4, k = i & 15;
  int dst = ei[NE + e];
  atomicAdd(&attrsum[dst * EA + k], eattr[i]);
  if (k == 0) atomicAdd(&deg[dst], 1.f);
}

__global__ void k_loop_fin(const float* __restrict__ deg, float* attrsum) {
  int i = blockIdx.x * 256 + threadIdx.x;
  if (i >= NN * EA) return;
  attrsum[i] = attrsum[i] / fmaxf(deg[i >> 4], 1.f);
}

// ---- BN + leaky -> f16 feature buffer -------------------------------------
__global__ void k_bn_in(const float* __restrict__ x,
                        const float* __restrict__ g, const float* __restrict__ b,
                        const float* __restrict__ mu, const float* __restrict__ var,
                        _Float16* __restrict__ h16) {
  int i = blockIdx.x * 256 + threadIdx.x;
  if (i >= NN * F) return;
  int c = i & (F - 1);
  float v = (x[i] - mu[c]) * rsqrtf(var[c] + 1e-5f) * g[c] + b[c];
  h16[i] = (_Float16)lrelu(v, 0.1f);
}

// ---- WMMA node linear: xl = h@Wl, xr = h@Wr -------------------------------
__global__ __launch_bounds__(128) void k_linear_wmma(
    const _Float16* __restrict__ h16,
    const float* __restrict__ Wl, const float* __restrict__ Wr,
    float* __restrict__ xl, float* __restrict__ xr) {
  // weights transposed into LDS as f16: sW[m][col*128 + k]
  __shared__ _Float16 sW[2][F * F];
  int t = threadIdx.x;
  for (int i = t; i < F * F; i += 128) {
    int k = i >> 7, col = i & (F - 1);
    sW[0][col * F + k] = (_Float16)Wl[i];
    sW[1][col * F + k] = (_Float16)Wr[i];
  }
  __syncthreads();

  int wave = t >> 5, lane = t & 31;
  int tile = blockIdx.x * 4 + wave;
  if (tile >= MTILES) return;          // uniform per wave -> EXEC all-ones in WMMA
  int m0   = tile * 16;
  int row  = m0 + (lane & 15);
  int koff = (lane >> 4) * 8;          // A-layout K split per ISA 7.12.2

  v16h a[4];
  #pragma unroll
  for (int kt = 0; kt < 4; ++kt) {
    const _Float16* p = h16 + row * F + kt * 32 + koff;
    #pragma unroll
    for (int j = 0; j < 8; ++j) { a[kt][j] = p[j]; a[kt][8 + j] = p[16 + j]; }
  }

  for (int mat = 0; mat < 2; ++mat) {
    const _Float16* W = sW[mat];
    float* out = mat ? xr : xl;
    #pragma unroll
    for (int nt = 0; nt < 8; ++nt) {
      int col = nt * 16 + (lane & 15);
      v8f c = {};
      #pragma unroll
      for (int kt = 0; kt < 4; ++kt) {
        v16h bfrag;
        const _Float16* q = W + col * F + kt * 32 + (lane >> 4) * 8;
        #pragma unroll
        for (int j = 0; j < 8; ++j) { bfrag[j] = q[j]; bfrag[8 + j] = q[16 + j]; }
        c = __builtin_amdgcn_wmma_f32_16x16x32_f16(false, a[kt], false, bfrag,
                                                   (short)0, c, false, false);
      }
      int orow = m0 + (lane >> 4) * 8;
      #pragma unroll
      for (int v = 0; v < 8; ++v) out[(orow + v) * F + col] = c[v];
    }
  }
}

// ---- per-layer init --------------------------------------------------------
__global__ void k_init_layer(float* agg, float* smax, float* denom) {
  int i = blockIdx.x * 256 + threadIdx.x;
  if (i < NN * F) agg[i] = 0.f;
  if (i < NN * 2) { smax[i] = -3.402823466e38f; denom[i] = 0.f; }
}

// ---- GATv2 scores: att . leaky0.2(xl[src]+xr[dst]+We@attr) ----------------
__global__ __launch_bounds__(256) void k_score(
    const int* __restrict__ ei, const float* __restrict__ eattr,
    const float* __restrict__ loop_attr,
    const float* __restrict__ xl, const float* __restrict__ xr,
    const float* __restrict__ We, const float* __restrict__ att,
    float* __restrict__ score, float* smax) {
  __shared__ float sWe[EA * F];
  __shared__ float sAtt[F];
  int t = threadIdx.x;
  for (int i = t; i < EA * F; i += 256) sWe[i] = We[i];
  if (t < F) sAtt[t] = att[t];
  __syncthreads();

  int idx = blockIdx.x * 256 + t;
  if (idx >= EE * 2) return;
  int e = idx >> 1, h = idx & 1;
  int src, dst; const float* ap;
  if (e < NE) { src = ei[e]; dst = ei[NE + e]; ap = eattr + e * EA; }
  else        { src = dst = e - NE;            ap = loop_attr + (e - NE) * EA; }
  float ab[EA];
  #pragma unroll
  for (int k = 0; k < EA; ++k) ab[k] = ap[k];

  const float* pl = xl + src * F + h * 64;
  const float* pr = xr + dst * F + h * 64;
  float s = 0.f;
  for (int cc = 0; cc < 64; ++cc) {
    float ee = 0.f;
    #pragma unroll
    for (int k = 0; k < EA; ++k) ee += ab[k] * sWe[k * F + h * 64 + cc];
    float m = pl[cc] + pr[cc] + ee;
    s += sAtt[h * 64 + cc] * lrelu(m, 0.2f);
  }
  score[idx] = s;
  atomicMaxF(&smax[dst * 2 + h], s);
}

// ---- exp + denominator (overwrites score with ex) -------------------------
__global__ void k_expsum(const int* __restrict__ ei, float* score,
                         const float* __restrict__ smax, float* denom) {
  int idx = blockIdx.x * 256 + threadIdx.x;
  if (idx >= EE * 2) return;
  int e = idx >> 1, h = idx & 1;
  int dst = (e < NE) ? ei[NE + e] : (e - NE);
  float ex = expf(score[idx] - smax[dst * 2 + h]);
  score[idx] = ex;
  atomicAdd(&denom[dst * 2 + h], ex);
}

// ---- aggregation: agg[dst] += alpha * xl[src] -----------------------------
__global__ void k_agg(const int* __restrict__ ei, const float* __restrict__ xl,
                      const float* __restrict__ ex, const float* __restrict__ denom,
                      float* agg) {
  int idx = blockIdx.x * 256 + threadIdx.x;
  if (idx >= EE * F) return;
  int e = idx >> 7, c = idx & (F - 1), h = c >> 6;
  int src, dst;
  if (e < NE) { src = ei[e]; dst = ei[NE + e]; } else { src = dst = e - NE; }
  float alpha = ex[e * 2 + h] / denom[dst * 2 + h];
  atomicAdd(&agg[dst * F + c], alpha * xl[src * F + c]);
}

// ---- layer-1 epilogue: +b1, leaky, BN2, leaky -> f16 ----------------------
__global__ void k_post_mid(const float* __restrict__ agg, const float* __restrict__ bias,
                           const float* __restrict__ g, const float* __restrict__ b,
                           const float* __restrict__ mu, const float* __restrict__ var,
                           _Float16* __restrict__ h16) {
  int i = blockIdx.x * 256 + threadIdx.x;
  if (i >= NN * F) return;
  int c = i & (F - 1);
  float v = lrelu(agg[i] + bias[c], 0.1f);
  v = (v - mu[c]) * rsqrtf(var[c] + 1e-5f) * g[c] + b[c];
  h16[i] = (_Float16)lrelu(v, 0.1f);
}

// ---- final epilogue: +b2, leaky -> out ------------------------------------
__global__ void k_post_out(const float* __restrict__ agg, const float* __restrict__ bias,
                           float* __restrict__ out) {
  int i = blockIdx.x * 256 + threadIdx.x;
  if (i >= NN * F) return;
  out[i] = lrelu(agg[i] + bias[i & (F - 1)], 0.1f);
}

// ---- copy edge_index bits into output tail --------------------------------
__global__ void k_copy_ei(const int* __restrict__ ei, int* __restrict__ out_tail) {
  int i = blockIdx.x * 256 + threadIdx.x;
  if (i < 2 * NE) out_tail[i] = ei[i];
}

static inline unsigned gsz(long long n) { return (unsigned)((n + 255) / 256); }

extern "C" void kernel_launch(void* const* d_in, const int* in_sizes, int n_in,
                              void* d_out, int out_size, void* d_ws, size_t ws_size,
                              hipStream_t stream) {
  const float* x      = (const float*)d_in[0];
  const int*   ei     = (const int*)  d_in[1];
  const float* eattr  = (const float*)d_in[2];
  const float* bn1g   = (const float*)d_in[3];
  const float* bn1b   = (const float*)d_in[4];
  const float* bn1m   = (const float*)d_in[5];
  const float* bn1v   = (const float*)d_in[6];
  const float* W1l    = (const float*)d_in[7];
  const float* W1r    = (const float*)d_in[8];
  const float* W1e    = (const float*)d_in[9];
  const float* att1   = (const float*)d_in[10];
  const float* b1     = (const float*)d_in[11];
  const float* bn2g   = (const float*)d_in[12];
  const float* bn2b   = (const float*)d_in[13];
  const float* bn2m   = (const float*)d_in[14];
  const float* bn2v   = (const float*)d_in[15];
  const float* W2l    = (const float*)d_in[16];
  const float* W2r    = (const float*)d_in[17];
  const float* W2e    = (const float*)d_in[18];
  const float* att2   = (const float*)d_in[19];
  const float* b2     = (const float*)d_in[20];
  float* out = (float*)d_out;

  char* w = (char*)d_ws;
  size_t off = 0;
  auto carve = [&](size_t bytes) -> void* {
    void* p = w + off;
    off = (off + bytes + 255) & ~(size_t)255;
    return p;
  };
  _Float16* h16   = (_Float16*)carve((size_t)NN * F * 2);
  float* xl       = (float*)carve((size_t)NN * F * 4);
  float* xr       = (float*)carve((size_t)NN * F * 4);
  float* score    = (float*)carve((size_t)EE * 2 * 4);
  float* smax     = (float*)carve((size_t)NN * 2 * 4);
  float* denom    = (float*)carve((size_t)NN * 2 * 4);
  float* agg      = (float*)carve((size_t)NN * F * 4);
  float* deg      = (float*)carve((size_t)NN * 4);
  float* attrsum  = (float*)carve((size_t)NN * EA * 4);   // becomes loop_attr
  (void)ws_size; (void)n_in; (void)in_sizes; (void)out_size;

  // self-loop attrs
  k_prep_zero<<<gsz((long long)NN * EA), 256, 0, stream>>>(deg, attrsum);
  k_deg_attr <<<gsz((long long)NE * EA), 256, 0, stream>>>(ei, eattr, deg, attrsum);
  k_loop_fin <<<gsz((long long)NN * EA), 256, 0, stream>>>(deg, attrsum);

  // BN1 + leaky -> f16
  k_bn_in<<<gsz((long long)NN * F), 256, 0, stream>>>(x, bn1g, bn1b, bn1m, bn1v, h16);

  unsigned linBlocks = (MTILES + 3) / 4;

  // ---- layer 1 ----
  k_init_layer <<<gsz((long long)NN * F), 256, 0, stream>>>(agg, smax, denom);
  k_linear_wmma<<<linBlocks, 128, 0, stream>>>(h16, W1l, W1r, xl, xr);
  k_score      <<<gsz((long long)EE * 2), 256, 0, stream>>>(ei, eattr, attrsum, xl, xr, W1e, att1, score, smax);
  k_expsum     <<<gsz((long long)EE * 2), 256, 0, stream>>>(ei, score, smax, denom);
  k_agg        <<<gsz((long long)EE * F), 256, 0, stream>>>(ei, xl, score, denom, agg);
  k_post_mid   <<<gsz((long long)NN * F), 256, 0, stream>>>(agg, b1, bn2g, bn2b, bn2m, bn2v, h16);

  // ---- layer 2 ----
  k_init_layer <<<gsz((long long)NN * F), 256, 0, stream>>>(agg, smax, denom);
  k_linear_wmma<<<linBlocks, 128, 0, stream>>>(h16, W2l, W2r, xl, xr);
  k_score      <<<gsz((long long)EE * 2), 256, 0, stream>>>(ei, eattr, attrsum, xl, xr, W2e, att2, score, smax);
  k_expsum     <<<gsz((long long)EE * 2), 256, 0, stream>>>(ei, score, smax, denom);
  k_agg        <<<gsz((long long)EE * F), 256, 0, stream>>>(ei, xl, score, denom, agg);
  k_post_out   <<<gsz((long long)NN * F), 256, 0, stream>>>(agg, b2, out);

  // edge_index passthrough (bit copy into tail of d_out)
  k_copy_ei<<<gsz((long long)2 * NE), 256, 0, stream>>>(ei, (int*)(out + OUT_H));
}
